// SelfAttention_10548439679537
// MI455X (gfx1250) — matrix operand order repaired and checked by reading, not compiled, not run on hardware
//
#include <hip/hip_runtime.h>

// ---------------------------------------------------------------------------
// Self-attention for B=4, N=2048, D=512 (fp32 in/out), bf16 WMMA internally.
// ---------------------------------------------------------------------------

typedef __bf16 bf16;
typedef __attribute__((ext_vector_type(16))) __bf16 v16bf;
typedef __attribute__((ext_vector_type(8)))  float  v8f;
typedef __attribute__((ext_vector_type(4)))  __bf16 bh4;
typedef __attribute__((ext_vector_type(8)))  __bf16 v8bf;
typedef __attribute__((ext_vector_type(4)))  int    v4i;

#define BATCH   4
#define SEQ     2048
#define DIM     512
#define M_TOT   (BATCH * SEQ)   // 8192 rows for QKV GEMM

// ---- workspace layout (bytes) ----
#define OFF_XB  ((size_t)0)                         // x bf16 [8192][512]       8,388,608
#define OFF_WT  ((size_t)8388608)                   // Wt bf16 [3][512][512]    1,572,864 (transposed: [n][k])
#define OFF_QB  ((size_t)9961472)                   // Q  bf16 [8192][512]
#define OFF_KB  ((size_t)18350080)                  // K  bf16 [8192][512]
#define OFF_VT  ((size_t)26738688)                  // Vt bf16 [4][512][2048]   (V transposed per batch)
// total = 35,127,296 bytes

union Frag16 { uint4 u[2]; v16bf v; };
union Pack8  { v8bf v; uint4 u; };

__device__ __forceinline__ v16bf load_frag(const bf16* p0, const bf16* p1) {
    Frag16 f;
    f.u[0] = *(const uint4*)p0;
    f.u[1] = *(const uint4*)p1;
    return f.v;
}

__device__ __forceinline__ v8f wmma_bf16(v16bf a, v16bf b, v8f c) {
    return __builtin_amdgcn_wmma_f32_16x16x32_bf16(false, a, false, b, (short)0, c, false, false);
}

// ---------------------------------------------------------------------------
// CDNA5 async global->LDS copy (GLOBAL_LOAD_ASYNC_TO_LDS_B128, ASYNCcnt).
// ROCm 7.2 declares the builtin as (v4i addrspace(1)*, v4i addrspace(3)*,
// imm offset, imm cpol) -- confirmed by the round-3 diagnostic.
// ---------------------------------------------------------------------------
#if __has_builtin(__builtin_amdgcn_global_load_async_to_lds_b128)
#define ASYNC_VIA_BUILTIN 1
#else
#define ASYNC_VIA_BUILTIN 0
#endif

__device__ __forceinline__ void async_copy_b128(void* lds, const void* g) {
#if ASYNC_VIA_BUILTIN
    __builtin_amdgcn_global_load_async_to_lds_b128(
        (__attribute__((address_space(1))) v4i*)(void*)g,
        (__attribute__((address_space(3))) v4i*)lds, 0, 0);
#else
    // VDST holds the LDS byte offset (low 32 bits of the generic LDS address)
    unsigned loff = (unsigned)(unsigned long long)lds;
    asm volatile("global_load_async_to_lds_b128 %0, %1, off"
                 :: "v"(loff), "v"(g) : "memory");
#endif
}

__device__ __forceinline__ void wait_async0() {
#if __has_builtin(__builtin_amdgcn_s_wait_asynccnt)
    __builtin_amdgcn_s_wait_asynccnt(0);
#else
    asm volatile("s_wait_asynccnt 0x0" ::: "memory");
#endif
}

// ---------------------------------------------------------------------------
// Kernel 1: fp32 -> bf16 (vectorized x4)
// ---------------------------------------------------------------------------
__global__ __launch_bounds__(256) void k_convert_x(const float* __restrict__ x,
                                                   bf16* __restrict__ xb, int n4) {
    int i = blockIdx.x * 256 + threadIdx.x;
    if (i < n4) {
        float4 v = ((const float4*)x)[i];
        bh4 h;
        h.x = (bf16)v.x; h.y = (bf16)v.y; h.z = (bf16)v.z; h.w = (bf16)v.w;
        ((bh4*)xb)[i] = h;
    }
}

// ---------------------------------------------------------------------------
// Kernel 2: W (fp32 [k][n]) -> Wt (bf16 [n][k]), one weight per blockIdx.y
// ---------------------------------------------------------------------------
__global__ __launch_bounds__(256) void k_convert_w(const float* __restrict__ Wq,
                                                   const float* __restrict__ Wk,
                                                   const float* __restrict__ Wv,
                                                   bf16* __restrict__ wt) {
    int idx = blockIdx.x * 256 + threadIdx.x;      // 0 .. 512*512-1
    int which = blockIdx.y;
    const float* W = (which == 0) ? Wq : ((which == 1) ? Wk : Wv);
    int k = idx >> 9;          // row of W (input dim)
    int n = idx & 511;         // col of W (atten dim)
    wt[(size_t)which * (DIM * DIM) + (size_t)n * DIM + k] = (bf16)W[idx];
}

// ---------------------------------------------------------------------------
// Kernel 3: QKV GEMM.  out[m,n] = sum_k xb[m,k] * W[k,n] + bias[n]
//   blockIdx.z: 0 -> Q (row-major), 1 -> K (row-major), 2 -> V (transposed Vt)
//   128x128 block, BK=32, double-buffered LDS fed by async copies
// ---------------------------------------------------------------------------
#define BM 128
#define BN 128
#define BK 32

__global__ __launch_bounds__(256) void k_qkv_gemm(const bf16* __restrict__ xb,
                                                  const bf16* __restrict__ wt,
                                                  const float* __restrict__ bq,
                                                  const float* __restrict__ bk,
                                                  const float* __restrict__ bv,
                                                  bf16* __restrict__ Qb,
                                                  bf16* __restrict__ Kb,
                                                  bf16* __restrict__ Vt) {
    __shared__ bf16 As[2][BM * BK];   // [m][k]
    __shared__ bf16 Bs[2][BN * BK];   // [n][k]

    const int t    = threadIdx.x;
    const int wave = t >> 5;
    const int lane = t & 31;
    const int wm   = wave >> 1;            // 0..3
    const int wn   = wave & 1;             // 0..1
    const int hi   = lane >> 4;            // lane-half
    const int ln   = lane & 15;

    const int m0 = blockIdx.x * BM;
    const int n0 = blockIdx.y * BN;
    const int which = blockIdx.z;
    const bf16*  W    = wt + (size_t)which * (DIM * DIM);
    const float* bias = (which == 0) ? bq : ((which == 1) ? bk : bv);

    v8f acc[2][4];
    #pragma unroll
    for (int mt = 0; mt < 2; ++mt)
        #pragma unroll
        for (int nt = 0; nt < 4; ++nt) acc[mt][nt] = 0;

    const int ldrow = t >> 1;              // 0..127
    const int ldk   = (t & 1) * 16;        // 0 or 16
    const bf16* aSrc = xb + (size_t)(m0 + ldrow) * DIM + ldk;
    const bf16* bSrc = W  + (size_t)(n0 + ldrow) * DIM + ldk;
    const int   lofs = ldrow * BK + ldk;

    // prologue: async-stage tile 0 into buffer 0
    async_copy_b128(&As[0][lofs],     aSrc);
    async_copy_b128(&As[0][lofs + 8], aSrc + 8);
    async_copy_b128(&Bs[0][lofs],     bSrc);
    async_copy_b128(&Bs[0][lofs + 8], bSrc + 8);

    int cur = 0;
    #pragma unroll 1
    for (int k0 = 0; k0 < DIM; k0 += BK) {
        wait_async0();          // this wave's copies for buffer `cur` have landed
        __syncthreads();        // ... and everyone else's too

        // kick off the next tile into the other buffer (overlaps the WMMAs)
        if (k0 + BK < DIM) {
            const bf16* as = aSrc + k0 + BK;
            const bf16* bs = bSrc + k0 + BK;
            async_copy_b128(&As[cur ^ 1][lofs],     as);
            async_copy_b128(&As[cur ^ 1][lofs + 8], as + 8);
            async_copy_b128(&Bs[cur ^ 1][lofs],     bs);
            async_copy_b128(&Bs[cur ^ 1][lofs + 8], bs + 8);
        }

        v16bf af[2];
        #pragma unroll
        for (int mt = 0; mt < 2; ++mt) {
            const bf16* p = &As[cur][(wm * 32 + mt * 16 + ln) * BK + hi * 16];
            af[mt] = load_frag(p, p + 8);
        }
        v16bf bfm[4];
        #pragma unroll
        for (int nt = 0; nt < 4; ++nt) {
            const bf16* p = &Bs[cur][(wn * 64 + nt * 16 + ln) * BK + hi * 16];
            bfm[nt] = load_frag(p, p + 8);
        }
        #pragma unroll
        for (int mt = 0; mt < 2; ++mt)
            #pragma unroll
            for (int nt = 0; nt < 4; ++nt)
                acc[mt][nt] = wmma_bf16(af[mt], bfm[nt], acc[mt][nt]);

        cur ^= 1;
    }

    // ---- epilogue: uniform branch hoisted out of the store loops ----
    if (which == 2) {
        // Vt[b][d][tok]: a lane's 8 accumulator rows are 8 consecutive tokens
        #pragma unroll
        for (int mt = 0; mt < 2; ++mt) {
            const int mrow0 = m0 + wm * 32 + mt * 16 + hi * 8;   // first of 8 rows
            const int bb    = mrow0 >> 11;
            const int tok0  = mrow0 & (SEQ - 1);
            #pragma unroll
            for (int nt = 0; nt < 4; ++nt) {
                const int ncol = n0 + wn * 64 + nt * 16 + ln;
                const float bval = bias[ncol];
                Pack8 pk;
                #pragma unroll
                for (int r = 0; r < 8; ++r)
                    pk.v[r] = (bf16)(acc[mt][nt][r] + bval);
                *(uint4*)(Vt + (size_t)bb * DIM * SEQ + (size_t)ncol * SEQ + tok0) = pk.u;
            }
        }
    } else {
        bf16* dst = (which == 0) ? Qb : Kb;
        #pragma unroll
        for (int mt = 0; mt < 2; ++mt) {
            #pragma unroll
            for (int nt = 0; nt < 4; ++nt) {
                const int ncol = n0 + wn * 64 + nt * 16 + ln;
                const float bval = bias[ncol];
                #pragma unroll
                for (int r = 0; r < 8; ++r) {
                    const int mrow = m0 + wm * 32 + mt * 16 + r + hi * 8;
                    dst[(size_t)mrow * DIM + ncol] = (bf16)(acc[mt][nt][r] + bval);
                }
            }
        }
    }
}

// ---------------------------------------------------------------------------
// Kernel 4: attention for one 16-query tile per workgroup (8 waves).
//   Pass 1: S[16][2048] = Q Kt  (WMMA, waves split key tiles; Q in registers)
//   Pass 2: row softmax in LDS -> unnormalized P (bf16) + rowsum
//   Pass 3: O = P V (WMMA, waves split 64 output cols each), scale 1/rowsum
// ---------------------------------------------------------------------------
#define ATT_SMEM (16*512*2 + 16*2048*4 + 16*2048*2 + 16*16*4 + 16*4)

__global__ __launch_bounds__(256) void k_attention(const bf16* __restrict__ Qb,
                                                   const bf16* __restrict__ Kb,
                                                   const bf16* __restrict__ Vt,
                                                   float* __restrict__ out) {
    extern __shared__ char smem[];
    bf16*  Qs     = (bf16*)smem;                                         // 16 x 512
    float* Ss     = (float*)(smem + 16*512*2);                           // 16 x 2048
    bf16*  Ps     = (bf16*)(smem + 16*512*2 + 16*2048*4);                // 16 x 2048
    float* red    = (float*)(smem + 16*512*2 + 16*2048*4 + 16*2048*2);   // 16 x 16
    float* rowsum = red + 256;                                           // 16

    const int t    = threadIdx.x;
    const int wave = t >> 5;
    const int lane = t & 31;
    const int hi   = lane >> 4;
    const int ln   = lane & 15;

    const int q0 = blockIdx.x * 16;
    const int b  = blockIdx.y;

    const bf16* Qg = Qb + ((size_t)b * SEQ + q0) * DIM;
    const bf16* Kg = Kb + (size_t)b * SEQ * DIM;
    const bf16* Vg = Vt + (size_t)b * DIM * SEQ;

    // async-stage Q tile: 16x512 bf16 = 8192 elems, 32 elems per thread
    {
        const int row = t >> 4;
        const int c0  = (t & 15) * 32;
        #pragma unroll
        for (int p = 0; p < 4; ++p)
            async_copy_b128(Qs + row * DIM + c0 + p * 8,
                            Qg + (size_t)row * DIM + c0 + p * 8);
    }
    wait_async0();
    __syncthreads();

    // hoist all 16 Q fragments into registers (invariant over the key loop)
    v16bf qf[16];
    #pragma unroll
    for (int d = 0; d < 16; ++d) {
        const bf16* pa = Qs + ln * DIM + d * 32 + hi * 16;
        qf[d] = load_frag(pa, pa + 8);
    }

    // ---- Pass 1: scores (two independent WMMA chains per tile) ----
    #pragma unroll 1
    for (int j = wave; j < SEQ / 16; j += 8) {
        const int key0 = j * 16;
        const bf16* bBase = Kg + (size_t)(key0 + ln) * DIM;
        __builtin_prefetch(bBase + (size_t)128 * DIM, 0, 0);   // this wave's next tile
        v8f acc0 = 0, acc1 = 0;
        #pragma unroll
        for (int d = 0; d < 16; d += 2) {
            const bf16* pb0 = bBase + (d + 0) * 32 + hi * 16;
            const bf16* pb1 = bBase + (d + 1) * 32 + hi * 16;
            v16bf k0 = load_frag(pb0, pb0 + 8);
            v16bf k1 = load_frag(pb1, pb1 + 8);
            acc0 = wmma_bf16(qf[d + 0], k0, acc0);
            acc1 = wmma_bf16(qf[d + 1], k1, acc1);
        }
        v8f acc = acc0 + acc1;
        #pragma unroll
        for (int r = 0; r < 8; ++r) {
            const int row = r + hi * 8;
            Ss[row * SEQ + key0 + ln] = acc[r];
        }
    }
    __syncthreads();

    // ---- Pass 2: softmax (P kept unnormalized, rowsum saved) ----
    {
        const int r = t & 15;          // query row
        const int c = t >> 4;          // 128-col chunk
        const float4* srow = (const float4*)(Ss + r * SEQ + c * 128);

        float m = -3.0e38f;
        #pragma unroll
        for (int i = 0; i < 32; ++i) {
            float4 v = srow[i];
            m = fmaxf(m, fmaxf(fmaxf(v.x, v.y), fmaxf(v.z, v.w)));
        }
        red[r * 16 + c] = m;
        __syncthreads();
        float rm = red[r * 16];
        #pragma unroll
        for (int i = 1; i < 16; ++i) rm = fmaxf(rm, red[r * 16 + i]);
        __syncthreads();

        float s = 0.0f;
        bh4* prow = (bh4*)(Ps + r * SEQ + c * 128);
        #pragma unroll
        for (int i = 0; i < 32; ++i) {
            float4 v = srow[i];
            float e0 = __expf(v.x - rm);
            float e1 = __expf(v.y - rm);
            float e2 = __expf(v.z - rm);
            float e3 = __expf(v.w - rm);
            s += (e0 + e1) + (e2 + e3);
            bh4 p;
            p.x = (bf16)e0; p.y = (bf16)e1; p.z = (bf16)e2; p.w = (bf16)e3;
            prow[i] = p;
        }
        red[r * 16 + c] = s;
        __syncthreads();
        float rs = 0.0f;
        #pragma unroll
        for (int i = 0; i < 16; ++i) rs += red[r * 16 + i];
        if (c == 0) rowsum[r] = rs;
        __syncthreads();
    }

    // ---- Pass 3: O = P @ V, each wave owns 64 output columns ----
    const int dbase = wave * 64;
    v8f oacc[4];
    #pragma unroll
    for (int nt = 0; nt < 4; ++nt) oacc[nt] = 0;

    #pragma unroll 1
    for (int k0 = 0; k0 < SEQ; k0 += 32) {
        const bf16* pa = Ps + ln * SEQ + k0 + hi * 16;
        v16bf a = load_frag(pa, pa + 8);
        #pragma unroll
        for (int nt = 0; nt < 4; ++nt) {
            const bf16* pb = Vg + (size_t)(dbase + nt * 16 + ln) * SEQ + k0 + hi * 16;
            v16bf vf = load_frag(pb, pb + 8);
            oacc[nt] = wmma_bf16(a, vf, oacc[nt]);
        }
    }

    float inv[8];
    #pragma unroll
    for (int r = 0; r < 8; ++r) inv[r] = 1.0f / rowsum[r + hi * 8];

    float* outg = out + ((size_t)b * SEQ + q0) * DIM;
    #pragma unroll
    for (int nt = 0; nt < 4; ++nt) {
        const int dcol = dbase + nt * 16 + ln;
        #pragma unroll
        for (int r = 0; r < 8; ++r) {
            const int row = r + hi * 8;
            outg[(size_t)row * DIM + dcol] = oacc[nt][r] * inv[r];
        }
    }
}

// ---------------------------------------------------------------------------
// Launch
// ---------------------------------------------------------------------------
extern "C" void kernel_launch(void* const* d_in, const int* in_sizes, int n_in,
                              void* d_out, int out_size, void* d_ws, size_t ws_size,
                              hipStream_t stream) {
    const float* x  = (const float*)d_in[0];
    const float* Wq = (const float*)d_in[1];
    const float* bq = (const float*)d_in[2];
    const float* Wk = (const float*)d_in[3];
    const float* bk = (const float*)d_in[4];
    const float* Wv = (const float*)d_in[5];
    const float* bv = (const float*)d_in[6];
    float* out = (float*)d_out;

    char* ws = (char*)d_ws;
    bf16* xb = (bf16*)(ws + OFF_XB);
    bf16* wt = (bf16*)(ws + OFF_WT);
    bf16* Qb = (bf16*)(ws + OFF_QB);
    bf16* Kb = (bf16*)(ws + OFF_KB);
    bf16* Vt = (bf16*)(ws + OFF_VT);

    // 1) x -> bf16 (4M elems, 4 per thread)
    const int n4 = (M_TOT * DIM) / 4;
    k_convert_x<<<dim3(n4 / 256), dim3(256), 0, stream>>>(x, xb, n4);

    // 2) W -> bf16 transposed [n][k]
    k_convert_w<<<dim3((DIM * DIM) / 256, 3), dim3(256), 0, stream>>>(Wq, Wk, Wv, wt);

    // 3) QKV GEMM (z: 0=Q, 1=K, 2=V-transposed)
    k_qkv_gemm<<<dim3(M_TOT / BM, DIM / BN, 3), dim3(256), 0, stream>>>(
        xb, wt, bq, bk, bv, Qb, Kb, Vt);

    // 4) attention: one 16-query tile per block
    k_attention<<<dim3(SEQ / 16, BATCH), dim3(256), ATT_SMEM, stream>>>(
        Qb, Kb, Vt, out);
}
